// Seq2SeqLSTMForecaster_86492051407468
// MI455X (gfx1250) — compile-verified, matrix operand order checked by reading, not run on hardware
//
#include <hip/hip_runtime.h>
#include <hip/hip_bf16.h>
#include <stdint.h>

typedef __attribute__((ext_vector_type(16))) _Float16 v16h;
typedef __attribute__((ext_vector_type(8)))  float    v8f;

// Global-address-space views so weight streams lower to global_load_b128
// (not flat_load_b128, which double-counts into LOADcnt+DScnt).
typedef __attribute__((address_space(1))) const _Float16 g_f16;
typedef __attribute__((address_space(1))) const v16h     g_v16h;

#define Hn   256
#define Bn   256
#define Sn   672
#define Tn   96
#define BT   16        // batch rows per workgroup
#define FRAG 512       // halves per 16x32 fragment block (32 lanes x 16 halves)

__device__ __forceinline__ const g_f16* to_global(const _Float16* p) {
  return (const g_f16*)(uintptr_t)p;   // int-cast: global AS == generic address
}

__device__ __forceinline__ float sigf(float v) { return 1.0f / (1.0f + __expf(-v)); }
__device__ __forceinline__ float tanh_fast(float v) {
  v = fminf(fmaxf(v, -15.0f), 15.0f);
  float e = __expf(2.0f * v);
  return (e - 1.0f) / (e + 1.0f);
}

// Defeat LICM: make the (uniform, SGPR-resident) weight base pointer opaque per
// timestep so fragment loads can't be hoisted out of the loop (which previously
// made the compiler preload ~1500 VGPRs of weights and spill to scratch).
__device__ __forceinline__ void opaque(const g_f16*& p) {
  asm volatile("" : "+s"(p));
}

// gates(acc) += A(afrag) x W^T for this wave's 8 gate tiles.
// Wave w owns gate tiles nt = g*16 + 2w + hf  (g = i,f,g,o ; hf = 0,1),
// i.e. all four gates for hidden columns [32w, 32w+32).
__device__ __forceinline__ void mm_acc(const _Float16* __restrict__ afrag,
                                       const g_f16* __restrict__ wfrag,
                                       v8f acc[8], int wave, int lane) {
  #pragma unroll
  for (int kt = 0; kt < 8; ++kt) {
    v16h a = *(const v16h*)(afrag + kt * FRAG + lane * 16);
    #pragma unroll
    for (int g4 = 0; g4 < 4; ++g4) {
      #pragma unroll
      for (int hf = 0; hf < 2; ++hf) {
        int nt = g4 * 16 + wave * 2 + hf;
        v16h b = *(const g_v16h*)(wfrag + (nt * 8 + kt) * FRAG + lane * 16);
        acc[g4 * 2 + hf] = __builtin_amdgcn_wmma_f32_16x16x32_f16(
            false, a, false, b, (short)0, acc[g4 * 2 + hf], false, false);
      }
    }
  }
}

// One LSTM cell step for one layer over a 16-row batch tile.
// h_frag: state h in WMMA A-fragment layout (LDS).  c kept in registers.
// Scalar-input layers pass xvec (Wih column, [4H]) + xbuf ([BT] scalars);
// matrix-input layers pass hin_frag + Wih_frag instead.
__device__ __forceinline__ void lstm_step(
    _Float16* __restrict__ h_frag,
    const g_f16* __restrict__ Whh_frag,
    const _Float16* __restrict__ hin_frag,
    const g_f16* __restrict__ Wih_frag,
    const float* __restrict__ xvec,
    const float* __restrict__ xbuf,
    const float* __restrict__ bias,
    float creg[16], int wave, int lane) {
  v8f z = {0.f, 0.f, 0.f, 0.f, 0.f, 0.f, 0.f, 0.f};
  v8f acc[8];
  #pragma unroll
  for (int i = 0; i < 8; ++i) acc[i] = z;

  mm_acc(h_frag, Whh_frag, acc, wave, lane);
  if (Wih_frag) mm_acc(hin_frag, Wih_frag, acc, wave, lane);

  __syncthreads();  // all waves finished reading h_frag; gates final

  int n15  = lane & 15;
  int mgrp = (lane >> 4) * 8;
  int khi  = (lane >> 3) & 1;
  #pragma unroll
  for (int hf = 0; hf < 2; ++hf) {
    int n = wave * 32 + hf * 16 + n15;
    float bi = bias[n], bf = bias[Hn + n], bg = bias[2 * Hn + n], bo = bias[3 * Hn + n];
    float wi = 0.f, wf = 0.f, wg = 0.f, wo = 0.f;
    if (xvec) { wi = xvec[n]; wf = xvec[Hn + n]; wg = xvec[2 * Hn + n]; wo = xvec[3 * Hn + n]; }
    #pragma unroll
    for (int r = 0; r < 8; ++r) {
      int m = mgrp + r;
      float x  = xbuf ? xbuf[m] : 0.f;
      float gi = acc[0 + hf][r] + bi + x * wi;
      float gf = acc[2 + hf][r] + bf + x * wf;
      float gg = acc[4 + hf][r] + bg + x * wg;
      float go = acc[6 + hf][r] + bo + x * wo;
      float c  = sigf(gf) * creg[hf * 8 + r] + sigf(gi) * tanh_fast(gg);
      float h  = sigf(go) * tanh_fast(c);
      creg[hf * 8 + r] = c;
      // scatter new h into this wave's exclusive fragment slab (K-tile == wave)
      h_frag[wave * FRAG + (m + 16 * khi) * 16 + (lane & 7) + hf * 8] = (_Float16)h;
    }
  }
  __syncthreads();  // new h visible to all waves
}

__global__ void __launch_bounds__(256, 1)
seq2seq_lstm_kernel(
    const float* __restrict__ features,
    const float* __restrict__ enc_Wih0, const float* __restrict__ enc_b0,
    const float* __restrict__ enc_b1,
    const float* __restrict__ dec_Wih0, const float* __restrict__ dec_b0,
    const float* __restrict__ dec_b1,
    const float* __restrict__ out_b1, const float* __restrict__ out_W2,
    const float* __restrict__ out_b2,
    const _Float16* __restrict__ wWhh0e_in, const _Float16* __restrict__ wWih1e_in,
    const _Float16* __restrict__ wWhh1e_in,
    const _Float16* __restrict__ wWhh0d_in, const _Float16* __restrict__ wWih1d_in,
    const _Float16* __restrict__ wWhh1d_in, const _Float16* __restrict__ wO1_in,
    float* __restrict__ out) {
  __shared__ __align__(32) _Float16 sh0[BT * Hn];   // layer-0 h, fragment layout
  __shared__ __align__(32) _Float16 sh1[BT * Hn];   // layer-1 h, fragment layout
  __shared__ float shid[BT * Hn];                   // head hidden
  __shared__ float sx[BT];                          // per-step scalar inputs / preds
  __shared__ float sred[256];                       // head reduction scratch

  int tid  = threadIdx.x;
  int wave = tid >> 5, lane = tid & 31;
  int tile = blockIdx.x;

  for (int i = tid; i < BT * Hn; i += 256) { sh0[i] = (_Float16)0.f; sh1[i] = (_Float16)0.f; }
  float c0reg[16], c1reg[16];
  #pragma unroll
  for (int i = 0; i < 16; ++i) { c0reg[i] = 0.f; c1reg[i] = 0.f; }
  __syncthreads();

  // -------- encoder: 672 steps, 2 layers, step-synchronous --------
  {
    const g_f16* wWhh0e = to_global(wWhh0e_in);
    const g_f16* wWih1e = to_global(wWih1e_in);
    const g_f16* wWhh1e = to_global(wWhh1e_in);
    for (int t = 0; t < Sn; ++t) {
      opaque(wWhh0e); opaque(wWih1e); opaque(wWhh1e);  // keep weight loads in-loop
      if (tid < BT) sx[tid] = features[(size_t)(tile * BT + tid) * Sn + t];
      lstm_step(sh0, wWhh0e, nullptr, nullptr, enc_Wih0, sx, enc_b0, c0reg, wave, lane);
      lstm_step(sh1, wWhh1e, sh0, wWih1e, nullptr, nullptr, enc_b1, c1reg, wave, lane);
    }
  }

  // -------- autoregressive decoder: 96 steps --------
  if (tid < BT) sx[tid] = 0.f;   // x0 = zeros
  __syncthreads();
  {
    const g_f16* wWhh0d = to_global(wWhh0d_in);
    const g_f16* wWih1d = to_global(wWih1d_in);
    const g_f16* wWhh1d = to_global(wWhh1d_in);
    const g_f16* wO1    = to_global(wO1_in);
    for (int t = 0; t < Tn; ++t) {
      opaque(wWhh0d); opaque(wWih1d); opaque(wWhh1d); opaque(wO1);
      lstm_step(sh0, wWhh0d, nullptr, nullptr, dec_Wih0, sx, dec_b0, c0reg, wave, lane);
      lstm_step(sh1, wWhh1d, sh0, wWih1d, nullptr, nullptr, dec_b1, c1reg, wave, lane);

      // head: hid = relu(h1 @ W1^T + b1); wave w owns hid columns [32w, 32w+32)
      v8f z = {0.f, 0.f, 0.f, 0.f, 0.f, 0.f, 0.f, 0.f};
      v8f hacc0 = z, hacc1 = z;
      #pragma unroll
      for (int kt = 0; kt < 8; ++kt) {
        v16h a  = *(const v16h*)(sh1 + kt * FRAG + lane * 16);
        v16h b0 = *(const g_v16h*)(wO1 + ((wave * 2 + 0) * 8 + kt) * FRAG + lane * 16);
        hacc0 = __builtin_amdgcn_wmma_f32_16x16x32_f16(false, a, false, b0, (short)0, hacc0, false, false);
        v16h b1 = *(const g_v16h*)(wO1 + ((wave * 2 + 1) * 8 + kt) * FRAG + lane * 16);
        hacc1 = __builtin_amdgcn_wmma_f32_16x16x32_f16(false, a, false, b1, (short)0, hacc1, false, false);
      }
      int n15 = lane & 15, mgrp = (lane >> 4) * 8;
      {
        int n = (wave * 2 + 0) * 16 + n15; float bv = out_b1[n];
        #pragma unroll
        for (int r = 0; r < 8; ++r) shid[(mgrp + r) * Hn + n] = fmaxf(hacc0[r] + bv, 0.f);
      }
      {
        int n = (wave * 2 + 1) * 16 + n15; float bv = out_b1[n];
        #pragma unroll
        for (int r = 0; r < 8; ++r) shid[(mgrp + r) * Hn + n] = fmaxf(hacc1[r] + bv, 0.f);
      }
      __syncthreads();

      // pred[m] = hid[m,:] . out_W2 + b2  (two-stage LDS reduction)
      {
        int m = tid >> 4, j = tid & 15;
        float s = 0.f;
        #pragma unroll
        for (int ii = 0; ii < 16; ++ii) s += shid[m * Hn + j * 16 + ii] * out_W2[j * 16 + ii];
        sred[m * 16 + j] = s;
      }
      __syncthreads();
      if (tid < BT) {
        float s = out_b2[0];
        #pragma unroll
        for (int i = 0; i < 16; ++i) s += sred[tid * 16 + i];
        sx[tid] = s;                                        // next-step input
        out[(size_t)(tile * BT + tid) * Tn + t] = s;        // preds.T -> [B, T]
      }
      __syncthreads();
    }
  }
}

// Convert fp32 [rows x cols] row-major weights to fp16 fragment-major blocks:
// block = nt*(cols/32)+kt, within-block = lane*16+half, matching the
// 16-bit A/B operand VGPR layout for v_wmma_f32_16x16x32_f16 (wave32).
__global__ void convert_frag_kernel(const float* __restrict__ src,
                                    _Float16* __restrict__ dst,
                                    int rows, int cols) {
  int idx = blockIdx.x * blockDim.x + threadIdx.x;
  if (idx >= rows * cols) return;
  int ktc    = cols >> 5;
  int block  = idx >> 9;
  int within = idx & 511;
  int lane = within >> 4;
  int half = within & 15;
  int nt = block / ktc;
  int kt = block - nt * ktc;
  int n   = nt * 16 + (lane & 15);
  int khi = lane >> 4;
  int k   = kt * 32 + (half & 7) + khi * 8 + ((half & 8) ? 16 : 0);
  dst[idx] = (_Float16)src[n * cols + k];
}

extern "C" void kernel_launch(void* const* d_in, const int* in_sizes, int n_in,
                              void* d_out, int out_size, void* d_ws, size_t ws_size,
                              hipStream_t stream) {
  const float* features = (const float*)d_in[0];
  const float* enc_Wih0 = (const float*)d_in[1];
  const float* enc_Whh0 = (const float*)d_in[2];
  const float* enc_b0   = (const float*)d_in[3];
  const float* enc_Wih1 = (const float*)d_in[4];
  const float* enc_Whh1 = (const float*)d_in[5];
  const float* enc_b1   = (const float*)d_in[6];
  const float* dec_Wih0 = (const float*)d_in[7];
  const float* dec_Whh0 = (const float*)d_in[8];
  const float* dec_b0   = (const float*)d_in[9];
  const float* dec_Wih1 = (const float*)d_in[10];
  const float* dec_Whh1 = (const float*)d_in[11];
  const float* dec_b1   = (const float*)d_in[12];
  const float* out_W1   = (const float*)d_in[13];
  const float* out_b1   = (const float*)d_in[14];
  const float* out_W2   = (const float*)d_in[15];
  const float* out_b2   = (const float*)d_in[16];

  const int NW = 4 * Hn * Hn;  // 262144 halves per [4H x H] matrix
  _Float16* ws    = (_Float16*)d_ws;   // needs ~3.3 MB of workspace
  _Float16* wWhh0e = ws + 0 * (size_t)NW;
  _Float16* wWih1e = ws + 1 * (size_t)NW;
  _Float16* wWhh1e = ws + 2 * (size_t)NW;
  _Float16* wWhh0d = ws + 3 * (size_t)NW;
  _Float16* wWih1d = ws + 4 * (size_t)NW;
  _Float16* wWhh1d = ws + 5 * (size_t)NW;
  _Float16* wO1    = ws + 6 * (size_t)NW;  // [H x H] = 65536 halves

  convert_frag_kernel<<<NW / 256, 256, 0, stream>>>(enc_Whh0, wWhh0e, 4 * Hn, Hn);
  convert_frag_kernel<<<NW / 256, 256, 0, stream>>>(enc_Wih1, wWih1e, 4 * Hn, Hn);
  convert_frag_kernel<<<NW / 256, 256, 0, stream>>>(enc_Whh1, wWhh1e, 4 * Hn, Hn);
  convert_frag_kernel<<<NW / 256, 256, 0, stream>>>(dec_Whh0, wWhh0d, 4 * Hn, Hn);
  convert_frag_kernel<<<NW / 256, 256, 0, stream>>>(dec_Wih1, wWih1d, 4 * Hn, Hn);
  convert_frag_kernel<<<NW / 256, 256, 0, stream>>>(dec_Whh1, wWhh1d, 4 * Hn, Hn);
  convert_frag_kernel<<<(Hn * Hn) / 256, 256, 0, stream>>>(out_W1, wO1, Hn, Hn);

  seq2seq_lstm_kernel<<<Bn / BT, 256, 0, stream>>>(
      features, enc_Wih0, enc_b0, enc_b1, dec_Wih0, dec_b0, dec_b1,
      out_b1, out_W2, out_b2,
      wWhh0e, wWih1e, wWhh1e, wWhh0d, wWih1d, wWhh1d, wO1,
      (float*)d_out);
}
